// BoxHead_31834297598413
// MI455X (gfx1250) — matrix operand order repaired and checked by reading, not compiled, not run on hardware
//
#include <hip/hip_runtime.h>

typedef __bf16 bf16_t;
typedef __attribute__((ext_vector_type(16))) __bf16 v16bf;
typedef __attribute__((ext_vector_type(8)))  __bf16 v8bf;
typedef __attribute__((ext_vector_type(8)))  float   v8f;
typedef __attribute__((ext_vector_type(4)))  float   v4f;
typedef __attribute__((ext_vector_type(4)))  unsigned u32x4;
typedef __attribute__((ext_vector_type(8)))  int      i32x8;
typedef __attribute__((ext_vector_type(4)))  int      i32x4;

// ---- feature gates ----------------------------------------------------------
#if defined(__has_builtin)
#if __has_builtin(__builtin_amdgcn_cvt_pk_bf16_f32)
#define USE_CVT_PK_BF16 1
#endif
#if __has_builtin(__builtin_amdgcn_tensor_load_to_lds) && __has_builtin(__builtin_amdgcn_s_wait_tensorcnt)
#define TDM_ENABLE 1
#endif
#endif
#ifndef USE_CVT_PK_BF16
#define USE_CVT_PK_BF16 0
#endif
#ifndef TDM_ENABLE
#define TDM_ENABLE 0
#endif

__device__ __forceinline__ bf16_t f2bf(float f) { // RNE fallback, bit-exact
    unsigned u = __builtin_bit_cast(unsigned, f);
    unsigned r = u + 0x7FFFu + ((u >> 16) & 1u);
    unsigned short h = (unsigned short)(r >> 16);
    return __builtin_bit_cast(bf16_t, h);
}

__device__ __forceinline__ void store_bf16x2(bf16_t* p, float a, float b) {
#if USE_CVT_PK_BF16
    auto v = __builtin_amdgcn_cvt_pk_bf16_f32(a, b);   // v_cvt_pk_bf16_f32
    __builtin_memcpy(p, &v, 4);
#else
    p[0] = f2bf(a); p[1] = f2bf(b);
#endif
}

#define SHUF16(lo, hi) __builtin_shufflevector((lo), (hi), 0,1,2,3,4,5,6,7,8,9,10,11,12,13,14,15)

template <typename AT> struct ARegs;
template <> struct ARegs<float>  { v4f  v[4]; };
template <> struct ARegs<bf16_t> { v8bf v[2]; };

// C = relu(A @ W + bias), A:[M,K] (fp32 or bf16), W:[K,Nc] fp32, out bf16 [M,Nc]
// Tile 128x128x32, 8 waves, wave tile 64x32 (4x2 wmma 16x16x32 accs).
// Double-buffered LDS pipeline. For bf16 A (GEMM2), the A tile is staged by the
// Tensor Data Mover (tensor_load_to_lds + s_wait_tensorcnt) with LDS padding
// matching the 80B bank-rotated row layout; fp32 A falls back to reg staging.
template <typename AT>
__global__ __launch_bounds__(256)
void gemm_bias_relu_wmma(const AT* __restrict__ A, const float* __restrict__ W,
                         const float* __restrict__ bias, bf16_t* __restrict__ out,
                         int M, int K, int Nc) {
    constexpr int BM = 128, BN = 128, BK = 32, LDA = 40; // 80B rows: 16B-aligned, bank-rotating
    constexpr bool kUseTDM = (TDM_ENABLE != 0) && (sizeof(AT) == 2);
    __shared__ bf16_t Asl[2][BM * LDA];
    __shared__ bf16_t Bsl[2][BN * LDA];

    const int tid  = threadIdx.x;
    const int lane = tid & 31;
    const int wave = tid >> 5;
    const int lm = lane & 15, lh = lane >> 4;
    const int wm = (wave >> 2) * 64;
    const int wn = (wave & 3)  * 32;
    const int mb = blockIdx.y * BM;
    const int nb = blockIdx.x * BN;
    const int Mten = ((M + 127) >> 7) << 7;   // allocated tensor rows (Mpad)

    // B-staging decode (constant per thread)
    const int n_in = tid & 127;
    const int kqb  = tid >> 7;           // 0..1
    const int gn   = nb + n_in;

    auto fetchA = [&](int k0, ARegs<AT>& R) {
        if constexpr (sizeof(AT) == 4) {
#pragma unroll
            for (int c = 0; c < 4; ++c) {
                int chunk = tid + 256 * c, row = chunk >> 3, kq = chunk & 7;
                int grow = mb + row; if (grow >= M) grow = M - 1; // clamp: row-local, stores guarded
                R.v[c] = *(const v4f*)(A + (size_t)grow * K + (k0 + 4 * kq));
            }
        } else {
#pragma unroll
            for (int c = 0; c < 2; ++c) {
                int chunk = tid + 256 * c, row = chunk >> 2, kq = chunk & 3;
                int grow = mb + row; if (grow >= M) grow = M - 1;
                R.v[c] = *(const v8bf*)(A + (size_t)grow * K + (k0 + 8 * kq));
            }
        }
    };
    auto storeA = [&](bf16_t* buf, ARegs<AT>& R) {
        if constexpr (sizeof(AT) == 4) {
#pragma unroll
            for (int c = 0; c < 4; ++c) {
                int chunk = tid + 256 * c, row = chunk >> 3, kq = chunk & 7;
                bf16_t* p = &buf[row * LDA + 4 * kq];
                store_bf16x2(p,     R.v[c].x, R.v[c].y);
                store_bf16x2(p + 2, R.v[c].z, R.v[c].w);
            }
        } else {
#pragma unroll
            for (int c = 0; c < 2; ++c) {
                int chunk = tid + 256 * c, row = chunk >> 2, kq = chunk & 3;
                *(v8bf*)&buf[row * LDA + 8 * kq] = R.v[c];
            }
        }
    };
    // TDM: DMA A tile [128 x 32] bf16 -> LDS with 16B pad per 64B row (row stride 80B)
    auto tdmA = [&](int k0, int buf) {
#if TDM_ENABLE
        if (wave == 0) {
            unsigned lds = (unsigned)(uintptr_t)(&Asl[buf][0]);
            unsigned long long ga =
                (unsigned long long)(uintptr_t)(A + (size_t)mb * K + k0);
            u32x4 g0;
            g0.x = 1u;                                            // count=1, user mode
            g0.y = lds;                                           // lds_addr
            g0.z = (unsigned)(ga & 0xFFFFFFFFu);                  // global_addr[31:0]
            g0.w = (unsigned)((ga >> 32) & 0x1FFFFFFu)            // global_addr[56:32]
                 | (2u << 30);                                    // type=2 (image)
            i32x8 g1;
            g1[0] = (int)((1u << 16)     // data_size = 2B
                        | (1u << 20)     // pad_enable
                        | (3u << 22)     // pad_interval: 16 DWORDs (=64B row)
                        | (3u << 25));   // pad_amount: 4 DWORDs (=16B)
            g1[1] = (int)(((unsigned)K & 0xFFFFu) << 16);         // tensor_dim0[15:0]
            g1[2] = (int)((((unsigned)K >> 16) & 0xFFFFu)         // tensor_dim0[31:16]
                        | (((unsigned)Mten & 0xFFFFu) << 16));    // tensor_dim1[15:0]
            g1[3] = (int)((((unsigned)Mten >> 16) & 0xFFFFu)      // tensor_dim1[31:16]
                        | (32u << 16));                           // tile_dim0 = 32
            g1[4] = 128;                                          // tile_dim1=128, tile_dim2=0
            g1[5] = (int)(unsigned)K;                             // tensor_dim0_stride[31:0]
            g1[6] = 0;                                            // stride hi bits
            g1[7] = 0;
            i32x4 gz4 = (i32x4){0, 0, 0, 0};
            i32x8 gz8 = (i32x8){0, 0, 0, 0, 0, 0, 0, 0};
            __builtin_amdgcn_tensor_load_to_lds(g0, g1, gz4, gz4, gz8, 0);
        }
#else
        (void)k0; (void)buf;
#endif
    };
    auto fetchB = [&](int k0, float* f) {
#pragma unroll
        for (int c = 0; c < 4; ++c) {
            int kk = k0 + 4 * (kqb + 2 * c);
            f[4*c+0] = W[(size_t)(kk + 0) * Nc + gn];
            f[4*c+1] = W[(size_t)(kk + 1) * Nc + gn];
            f[4*c+2] = W[(size_t)(kk + 2) * Nc + gn];
            f[4*c+3] = W[(size_t)(kk + 3) * Nc + gn];
        }
    };
    auto storeB = [&](bf16_t* buf, const float* f) {
#pragma unroll
        for (int c = 0; c < 4; ++c) {
            bf16_t* p = &buf[n_in * LDA + 4 * (kqb + 2 * c)]; // transposed [n][k]
            store_bf16x2(p,     f[4*c+0], f[4*c+1]);
            store_bf16x2(p + 2, f[4*c+2], f[4*c+3]);
        }
    };

    v8f acc[4][2];
#pragma unroll
    for (int i = 0; i < 4; ++i)
#pragma unroll
        for (int j = 0; j < 2; ++j)
            acc[i][j] = (v8f){0.f,0.f,0.f,0.f,0.f,0.f,0.f,0.f};

    const int ksteps = K / BK;
    ARegs<AT> ra; float rb[16];
    if constexpr (kUseTDM) tdmA(0, 0); else fetchA(0, ra);
    fetchB(0, rb);
    if constexpr (!kUseTDM) storeA(Asl[0], ra);
    storeB(Bsl[0], rb);
#if TDM_ENABLE
    if constexpr (kUseTDM) __builtin_amdgcn_s_wait_tensorcnt(0);
#endif
    __syncthreads();

    for (int kt = 0; kt < ksteps; ++kt) {
        const int cur = kt & 1, nxt = cur ^ 1;
        const bool more = (kt + 1) < ksteps;
        if (more) {
            if constexpr (kUseTDM) tdmA((kt + 1) * BK, nxt);
            else fetchA((kt + 1) * BK, ra);
            fetchB((kt + 1) * BK, rb);
        }

        // ---- fragments per ISA lane layout + 8 WMMAs ----
        v16bf afr[4], bfr[2];
#pragma unroll
        for (int i = 0; i < 4; ++i) {
            const bf16_t* p = &Asl[cur][(wm + 16 * i + lm) * LDA + 8 * lh];
            v8bf lo = *(const v8bf*)p;          // K = 8*lh .. +7
            v8bf hi = *(const v8bf*)(p + 16);   // K = 16+8*lh .. +7
            afr[i] = SHUF16(lo, hi);
        }
#pragma unroll
        for (int j = 0; j < 2; ++j) {
            const bf16_t* p = &Bsl[cur][(wn + 16 * j + lm) * LDA + 8 * lh];
            v8bf lo = *(const v8bf*)p;          // K = 8*lh .. +7
            v8bf hi = *(const v8bf*)(p + 16);   // K = 16+8*lh .. +7
            bfr[j] = SHUF16(lo, hi);
        }
#pragma unroll
        for (int i = 0; i < 4; ++i)
#pragma unroll
            for (int j = 0; j < 2; ++j)
                acc[i][j] = __builtin_amdgcn_wmma_f32_16x16x32_bf16(
                    false, afr[i], false, bfr[j], (short)0, acc[i][j], false, false);

        if (more) {
            if constexpr (!kUseTDM) storeA(Asl[nxt], ra);
            storeB(Bsl[nxt], rb);
        }
#if TDM_ENABLE
        if constexpr (kUseTDM) __builtin_amdgcn_s_wait_tensorcnt(0);
#endif
        __syncthreads();
    }

    // ---- epilogue: bias + relu + bf16 store ----
#pragma unroll
    for (int j = 0; j < 2; ++j) {
        const int col = nb + wn + 16 * j + lm;
        const float bv = bias[col];
#pragma unroll
        for (int i = 0; i < 4; ++i) {
#pragma unroll
            for (int r = 0; r < 8; ++r) {
                int row = mb + wm + 16 * i + 8 * lh + r;
                if (row < M) {
                    float v = acc[i][j][r] + bv;
                    v = v > 0.f ? v : 0.f;
                    out[(size_t)row * Nc + col] = f2bf(v);
                }
            }
        }
    }
}

// heads: out[:,0:4] = h @ Wc + bc ; out[:,4:16] = h @ Wr + br (two flat fp32 outputs)
__global__ __launch_bounds__(256)
void head_wmma(const bf16_t* __restrict__ h, const float* __restrict__ Wc,
               const float* __restrict__ bc, const float* __restrict__ Wr,
               const float* __restrict__ br, float* __restrict__ out, int M) {
    constexpr int HH = 1024, LDW = HH + 8; // 2064B rows: 16B aligned, bank-rotating
    __shared__ bf16_t Wsl[16 * LDW];
    const int tid = threadIdx.x;
    for (int idx = tid; idx < 16 * (HH / 2); idx += 256) {
        int c = idx >> 9, k = (idx & 511) * 2;
        float v0 = (c < 4) ? Wc[(size_t)k * 4 + c]       : Wr[(size_t)k * 12 + (c - 4)];
        float v1 = (c < 4) ? Wc[(size_t)(k + 1) * 4 + c] : Wr[(size_t)(k + 1) * 12 + (c - 4)];
        store_bf16x2(&Wsl[c * LDW + k], v0, v1);
    }
    __syncthreads();

    const int lane = tid & 31, wave = tid >> 5;
    const int lm = lane & 15, lh = lane >> 4;
    const int rbase = blockIdx.x * 128 + wave * 16;
    int arow = rbase + lm; if (arow >= M) arow = M - 1;
    const bf16_t* ap = h + (size_t)arow * HH;

    v8f acc = (v8f){0.f,0.f,0.f,0.f,0.f,0.f,0.f,0.f};
#pragma unroll 4
    for (int kt = 0; kt < HH / 32; ++kt) {
        const int k0 = kt * 32;
        v8bf alo = *(const v8bf*)(ap + k0 + 8 * lh);
        v8bf ahi = *(const v8bf*)(ap + k0 + 16 + 8 * lh);
        v16bf afr = SHUF16(alo, ahi);
        const bf16_t* bp = &Wsl[lm * LDW + k0 + 8 * lh];
        v8bf blo = *(const v8bf*)bp;
        v8bf bhi = *(const v8bf*)(bp + 16);
        v16bf bfr = SHUF16(blo, bhi);
        acc = __builtin_amdgcn_wmma_f32_16x16x32_bf16(
            false, afr, false, bfr, (short)0, acc, false, false);
    }

    const int col = lm;
    const float bv = (col < 4) ? bc[col] : br[col - 4];
#pragma unroll
    for (int r = 0; r < 8; ++r) {
        int row = rbase + 8 * lh + r;
        if (row < M) {
            float v = acc[r] + bv;
            if (col < 4) out[(size_t)row * 4 + col] = v;
            else         out[(size_t)M * 4 + (size_t)row * 12 + (col - 4)] = v;
        }
    }
}

extern "C" void kernel_launch(void* const* d_in, const int* in_sizes, int n_in,
                              void* d_out, int out_size, void* d_ws, size_t ws_size,
                              hipStream_t stream) {
    const float* x  = (const float*)d_in[0];
    const float* W1 = (const float*)d_in[1];
    const float* b1 = (const float*)d_in[2];
    const float* W2 = (const float*)d_in[3];
    const float* b2 = (const float*)d_in[4];
    const float* Wc = (const float*)d_in[5];
    const float* bc = (const float*)d_in[6];
    const float* Wr = (const float*)d_in[7];
    const float* br = (const float*)d_in[8];

    const int Hh   = in_sizes[2];           // 1024
    const int Din  = in_sizes[1] / Hh;      // 12544
    const int M    = in_sizes[0] / Din;     // 20000
    const int Mpad = ((M + 127) / 128) * 128;

    bf16_t* h1 = (bf16_t*)d_ws;
    bf16_t* h2 = h1 + (size_t)Mpad * Hh;

    dim3 grid(Hh / 128, Mpad / 128);
    gemm_bias_relu_wmma<float ><<<grid, 256, 0, stream>>>(x,  W1, b1, h1, M, Din, Hh);
    gemm_bias_relu_wmma<bf16_t><<<grid, 256, 0, stream>>>(h1, W2, b2, h2, M, Hh,  Hh);
    head_wmma<<<Mpad / 128, 256, 0, stream>>>(h2, Wc, bc, Wr, br, (float*)d_out, M);
}
